// WeightedHAN_87162066305583
// MI455X (gfx1250) — compile-verified, matrix operand order checked by reading, not run on hardware
//
#include <hip/hip_runtime.h>
#include <hip/hip_bf16.h>
#include <math.h>

// ---- problem constants (from reference) ----
#define NS   20000
#define NT2  4096
#define NTOT 24096          // NS + NT2
#define HH   4
#define DD   32
#define FF   128
#define CC   128
#define MTILES (NTOT/16)    // 1506
#define MTILES_S (NS/16)    // 1250 (exact split: s tiles then t tiles)

typedef __attribute__((ext_vector_type(16))) _Float16 v16h;
typedef __attribute__((ext_vector_type(8)))  _Float16 v8h;
typedef __attribute__((ext_vector_type(8)))  float    v8f;

// ---- order-preserving float<->uint key for atomic max ----
__device__ __forceinline__ unsigned fkey(float f) {
  unsigned b = __float_as_uint(f);
  return (b & 0x80000000u) ? ~b : (b | 0x80000000u);
}
__device__ __forceinline__ float funkey(unsigned k) {
  unsigned b = (k & 0x80000000u) ? (k & 0x7FFFFFFFu) : ~k;
  return __uint_as_float(b);
}

// edge type: ('s','s')->0, ('t','t')->1, ('s','t')->2, ('t','s')->3
// stack slot within dest type: k in {0,1} -> slot 0, k in {2,3} -> slot 1
__device__ __forceinline__ void edge_type(int src, int dst, int& k, int& dl, int& slot) {
  bool ss = src < NS, dd = dst < NS;
  k = ss ? (dd ? 0 : 2) : (dd ? 3 : 1);
  dl = dd ? dst : dst - NS;
  slot = (k >= 2) ? 1 : 0;
}

// ---- WMMA 16x16 tile from f16 LDS tiles; K=128 via 4x k32 steps ----
// 16-bit A 16x32 lane map: lane L: row M=L&15, K = {kh..kh+7} u {kh+16..kh+23}, kh=8*(L>>4).
// B assumed symmetric with lane -> column N (B tile stored as WT[n][k], row-major in k).
__device__ __forceinline__ v8f wmma_tile(const _Float16* As, const _Float16* Bs, int lane) {
  int r = lane & 15;
  int khalf = (lane >> 4) << 3;
  const _Float16* ar = As + r * CC;
  const _Float16* br = Bs + r * CC;
  v8f acc = {};
#pragma unroll
  for (int kc = 0; kc < 4; ++kc) {
    int k0 = kc * 32 + khalf;
    v8h a0 = *(const v8h*)(ar + k0);
    v8h a1 = *(const v8h*)(ar + k0 + 16);
    v8h b0 = *(const v8h*)(br + k0);
    v8h b1 = *(const v8h*)(br + k0 + 16);
    v16h a = __builtin_shufflevector(a0, a1, 0,1,2,3,4,5,6,7,8,9,10,11,12,13,14,15);
    v16h b = __builtin_shufflevector(b0, b1, 0,1,2,3,4,5,6,7,8,9,10,11,12,13,14,15);
    acc = __builtin_amdgcn_wmma_f32_16x16x32_f16(false, a, false, b, (short)0, acc,
                                                 false, false);
  }
  return acc;
}

// ---- transpose + f32->f16 convert: WT[c][r] = (f16)W[r][c], 128x128 ----
__global__ void k_transpose_cvt(const float* __restrict__ W, _Float16* __restrict__ WT) {
  int idx = blockIdx.x * blockDim.x + threadIdx.x;
  if (idx >= CC * FF) return;
  int r = idx / FF, c = idx % FF;
  WT[c * CC + r] = (_Float16)W[r * FF + c];
}

// ---- elementwise f32 -> f16 ----
__global__ void k_cvt(const float* __restrict__ X, _Float16* __restrict__ Y, int n) {
  int i = blockIdx.x * blockDim.x + threadIdx.x;
  if (i < n) Y[i] = (_Float16)X[i];
}

// ---- projection: h = X @ {Ws|Wt} + {bs|bt}; one block per 16-row M tile, 8 waves = 8 N tiles ----
__global__ void __launch_bounds__(256)
k_project(const _Float16* __restrict__ Xh,
          const _Float16* __restrict__ WThs, const _Float16* __restrict__ WTht,
          const float* __restrict__ bs,  const float* __restrict__ bt,
          float* __restrict__ Hout) {
  __shared__ float4 AsV[16 * CC / 8];   // 4 KB  (16x128 f16)
  __shared__ float4 BsV[FF * CC / 8];   // 32 KB (128x128 f16, full transposed weight)
  int tm = blockIdx.x;
  bool isS = tm < MTILES_S;
  const _Float16* WT   = isS ? WThs : WTht;
  const float*    bias = isS ? bs : bt;
  int t = threadIdx.x;
  const float4* asrc = (const float4*)(Xh + (size_t)tm * 16 * CC);
  const float4* bsrc = (const float4*)WT;
  AsV[t] = asrc[t];                               // 256 float4 == 256 threads
#pragma unroll
  for (int i = 0; i < 8; ++i) BsV[t + 256 * i] = bsrc[t + 256 * i];
  __syncthreads();
  int wave = t >> 5, lane = t & 31;
  const _Float16* As = (const _Float16*)AsV;
  const _Float16* Bs = (const _Float16*)BsV + wave * 16 * CC;
  v8f acc = wmma_tile(As, Bs, lane);
  int n  = (wave << 4) + (lane & 15);
  float bv = bias[n];
  int mb = (tm << 4) + ((lane >> 4) << 3);
#pragma unroll
  for (int i = 0; i < 8; ++i)
    Hout[(size_t)(mb + i) * FF + n] = acc[i] + bv;
}

// ---- edge logits (leaky relu) + segment max ----
__global__ void k_edge_logits(const int* __restrict__ EI, const float* __restrict__ Hf,
                              const float* __restrict__ a_src, const float* __restrict__ a_dst,
                              float* __restrict__ logits, unsigned* __restrict__ segmax, int E) {
  int e = blockIdx.x * blockDim.x + threadIdx.x;
  if (e >= E) return;
  int src = EI[e], dst = EI[E + e];
  int k, dl, slot; edge_type(src, dst, k, dl, slot); (void)slot;
  const float* hs = Hf + (size_t)src * FF;   // h rows laid out s-then-t == global node index
  const float* hd = Hf + (size_t)dst * FF;
  __builtin_prefetch(hs, 0, 1);              // global_prefetch_b8: start both row fetches
  __builtin_prefetch(hd, 0, 1);
  const float* as = a_src + k * FF;
  const float* ad = a_dst + k * FF;
#pragma unroll
  for (int h = 0; h < HH; ++h) {
    float la = 0.f, ld = 0.f;
#pragma unroll
    for (int d = 0; d < DD; ++d) {
      la += hs[h * DD + d] * as[h * DD + d];
      ld += hd[h * DD + d] * ad[h * DD + d];
    }
    float lg = la + ld;
    lg = lg > 0.f ? lg : 0.2f * lg;            // leaky relu, slope 0.2
    logits[(size_t)e * HH + h] = lg;
    atomicMax(&segmax[((size_t)k * NTOT + dl) * HH + h], fkey(lg));
  }
}

// ---- exp(l - max) in place + segment sum ----
__global__ void k_edge_exp(const int* __restrict__ EI, float* __restrict__ logits,
                           const unsigned* __restrict__ segmax, float* __restrict__ segsum, int E) {
  int e = blockIdx.x * blockDim.x + threadIdx.x;
  if (e >= E) return;
  int src = EI[e], dst = EI[E + e];
  int k, dl, slot; edge_type(src, dst, k, dl, slot); (void)slot;
#pragma unroll
  for (int h = 0; h < HH; ++h) {
    float mx = funkey(segmax[((size_t)k * NTOT + dl) * HH + h]);
    float ev = __expf(logits[(size_t)e * HH + h] - mx);
    logits[(size_t)e * HH + h] = ev;
    atomicAdd(&segsum[((size_t)k * NTOT + dl) * HH + h], ev);
  }
}

// ---- weighted message scatter: one thread per (edge, feature) ----
__global__ void k_edge_scatter(const int* __restrict__ EI, const float* __restrict__ ew,
                               const float* __restrict__ Hf, const float* __restrict__ expv,
                               const float* __restrict__ segsum, float* __restrict__ stk, int E) {
  int idx = blockIdx.x * blockDim.x + threadIdx.x;   // E*FF = 16.7M < 2^31
  if (idx >= E * FF) return;
  int e = idx >> 7, f = idx & 127, h = f >> 5;
  int src = EI[e], dst = EI[E + e];
  int k, dl, slot; edge_type(src, dst, k, dl, slot);
  float ev   = expv[(size_t)e * HH + h];
  float ssum = segsum[((size_t)k * NTOT + dl) * HH + h];
  float alpha = ev / (ssum + 1e-16f) * ew[e];
  float val = Hf[(size_t)src * FF + f] * alpha;
  atomicAdd(&stk[((size_t)slot * NTOT + dst) * FF + f], val);
}

// ---- relu in place (f32) + f16 copy for the semantic GEMM ----
__global__ void k_relu_cvt(float* __restrict__ p, _Float16* __restrict__ ph, int n) {
  int i = blockIdx.x * blockDim.x + threadIdx.x;
  if (i >= n) return;
  float v = fmaxf(p[i], 0.f);
  p[i] = v;
  ph[i] = (_Float16)v;
}

// ---- semantic GEMM: tanh(stk@Wk+bk), accumulate column means per (dest-type, slot) ----
__global__ void __launch_bounds__(256)
k_semantic(const _Float16* __restrict__ stkh, const _Float16* __restrict__ WThk,
           const float* __restrict__ bk, float* __restrict__ colsum) {
  __shared__ float4 AsV[16 * CC / 8];   // 4 KB
  __shared__ float4 BsV[FF * CC / 8];   // 32 KB
  int slot = blockIdx.x / MTILES;
  int tm   = blockIdx.x % MTILES;
  int t = threadIdx.x;
  const float4* asrc = (const float4*)(stkh + ((size_t)slot * NTOT + (size_t)tm * 16) * FF);
  const float4* bsrc = (const float4*)WThk;
  AsV[t] = asrc[t];
#pragma unroll
  for (int i = 0; i < 8; ++i) BsV[t + 256 * i] = bsrc[t + 256 * i];
  __syncthreads();
  int wave = t >> 5, lane = t & 31;
  const _Float16* As = (const _Float16*)AsV;
  const _Float16* Bs = (const _Float16*)BsV + wave * 16 * CC;
  v8f acc = wmma_tile(As, Bs, lane);
  int n = (wave << 4) + (lane & 15);
  float bv = bk[n];
  int dt = (tm < MTILES_S) ? 0 : 1;
  float inv = (dt == 0) ? (1.f / NS) : (1.f / NT2);
  float p = 0.f;
#pragma unroll
  for (int i = 0; i < 8; ++i) p += tanhf(acc[i] + bv);
  atomicAdd(&colsum[(dt * 2 + slot) * FF + n], p * inv);
}

// ---- semantic scores: softmax over 2 slots per dest type ----
__global__ void k_scores(const float* __restrict__ colsum, const float* __restrict__ q,
                         float* __restrict__ attn) {
  if (threadIdx.x != 0 || blockIdx.x != 0) return;
  for (int dt = 0; dt < 2; ++dt) {
    float s0 = 0.f, s1 = 0.f;
    for (int f = 0; f < FF; ++f) {
      s0 += q[f] * colsum[(dt * 2 + 0) * FF + f];
      s1 += q[f] * colsum[(dt * 2 + 1) * FF + f];
    }
    float m = fmaxf(s0, s1);
    float e0 = __expf(s0 - m), e1 = __expf(s1 - m);
    float inv = 1.f / (e0 + e1);
    attn[dt * 2 + 0] = e0 * inv;
    attn[dt * 2 + 1] = e1 * inv;
  }
}

// ---- combine slots with semantic attention; optional inter-layer relu ----
__global__ void k_combine(const float* __restrict__ stk, const float* __restrict__ attn,
                          float* __restrict__ dst, int doRelu) {
  int idx = blockIdx.x * blockDim.x + threadIdx.x;
  if (idx >= NTOT * FF) return;
  int node = idx >> 7;
  int dt = node < NS ? 0 : 1;
  float v = attn[dt * 2 + 0] * stk[idx] + attn[dt * 2 + 1] * stk[NTOT * FF + idx];
  dst[idx] = doRelu ? fmaxf(v, 0.f) : v;
}

extern "C" void kernel_launch(void* const* d_in, const int* in_sizes, int n_in,
                              void* d_out, int out_size, void* d_ws, size_t ws_size,
                              hipStream_t stream) {
  const float* x   = (const float*)d_in[0];
  const int*   ei  = (const int*)d_in[1];
  const float* ew  = (const float*)d_in[2];
  const float* Ws  = (const float*)d_in[3];
  const float* bs  = (const float*)d_in[4];
  const float* Wt  = (const float*)d_in[5];
  const float* bt  = (const float*)d_in[6];
  const float* asr = (const float*)d_in[7];
  const float* ads = (const float*)d_in[8];
  const float* Wk  = (const float*)d_in[9];
  const float* bk  = (const float*)d_in[10];
  const float* q   = (const float*)d_in[11];
  float* out = (float*)d_out;

  const int B = in_sizes[0] / (NTOT * CC);
  const int E = in_sizes[2] / B;
  const int L = in_sizes[3] / (CC * FF);

  // workspace carve-out (~75 MB)
  char* w = (char*)d_ws;
  auto carve = [&](size_t bytes) { char* p = w; w += (bytes + 255) & ~(size_t)255; return p; };
  float*     hbuf   = (float*)carve((size_t)NTOT * FF * 4);
  float*     logits = (float*)carve((size_t)E * HH * 4);       // reused for exp values
  unsigned*  segmax = (unsigned*)carve((size_t)4 * NTOT * HH * 4);
  float*     segsum = (float*)carve((size_t)4 * NTOT * HH * 4);
  float*     stk    = (float*)carve((size_t)2 * NTOT * FF * 4);
  _Float16*  stkh   = (_Float16*)carve((size_t)2 * NTOT * FF * 2);
  _Float16*  Xh     = (_Float16*)carve((size_t)NTOT * CC * 2);
  _Float16*  WThs   = (_Float16*)carve((size_t)FF * CC * 2);
  _Float16*  WTht   = (_Float16*)carve((size_t)FF * CC * 2);
  _Float16*  WThk   = (_Float16*)carve((size_t)FF * CC * 2);
  float*     colsum = (float*)carve((size_t)4 * FF * 4);
  float*     attn   = (float*)carve((size_t)4 * 4);
  float*     act    = (float*)carve((size_t)NTOT * FF * 4);

  const int TB = 256;
  for (int b = 0; b < B; ++b) {
    for (int l = 0; l < L; ++l) {
      const float* Xin = (l == 0) ? (x + (size_t)b * NTOT * CC) : act;
      const float* Wsl = Ws + (size_t)l * CC * FF;
      const float* Wtl = Wt + (size_t)l * CC * FF;
      const float* Wkl = Wk + (size_t)l * FF * FF;
      const float* bsl = bs + (size_t)l * FF;
      const float* btl = bt + (size_t)l * FF;
      const float* bkl = bk + (size_t)l * FF;
      const float* ql  = q  + (size_t)l * FF;
      const float* asl = asr + (size_t)l * 4 * HH * DD;
      const float* adl = ads + (size_t)l * 4 * HH * DD;
      const int*   EIb = ei + (size_t)b * 2 * E;
      const float* ewb = ew + (size_t)b * E;

      k_transpose_cvt<<<(CC * FF + TB - 1) / TB, TB, 0, stream>>>(Wsl, WThs);
      k_transpose_cvt<<<(CC * FF + TB - 1) / TB, TB, 0, stream>>>(Wtl, WTht);
      k_transpose_cvt<<<(CC * FF + TB - 1) / TB, TB, 0, stream>>>(Wkl, WThk);
      k_cvt<<<(NTOT * CC + TB - 1) / TB, TB, 0, stream>>>(Xin, Xh, NTOT * CC);

      k_project<<<MTILES, 256, 0, stream>>>(Xh, WThs, WTht, bsl, btl, hbuf);

      hipMemsetAsync(segmax, 0, (size_t)4 * NTOT * HH * 4, stream);
      hipMemsetAsync(segsum, 0, (size_t)4 * NTOT * HH * 4, stream);
      hipMemsetAsync(stk,    0, (size_t)2 * NTOT * FF * 4, stream);
      hipMemsetAsync(colsum, 0, (size_t)4 * FF * 4, stream);

      k_edge_logits<<<(E + TB - 1) / TB, TB, 0, stream>>>(EIb, hbuf, asl, adl, logits, segmax, E);
      k_edge_exp<<<(E + TB - 1) / TB, TB, 0, stream>>>(EIb, logits, segmax, segsum, E);
      k_edge_scatter<<<(E * FF + TB - 1) / TB, TB, 0, stream>>>(EIb, ewb, hbuf, logits, segsum, stk, E);

      k_relu_cvt<<<(2 * NTOT * FF + TB - 1) / TB, TB, 0, stream>>>(stk, stkh, 2 * NTOT * FF);

      k_semantic<<<2 * MTILES, 256, 0, stream>>>(stkh, WThk, bkl, colsum);
      k_scores<<<1, 1, 0, stream>>>(colsum, ql, attn);

      float* dstp = (l == L - 1) ? (out + (size_t)b * NTOT * FF) : act;
      k_combine<<<(NTOT * FF + TB - 1) / TB, TB, 0, stream>>>(stk, attn, dstp, (l < L - 1) ? 1 : 0);
    }
  }
}
